// GNN_8830452760601
// MI455X (gfx1250) — compile-verified, matrix operand order checked by reading, not run on hardware
//
#include <hip/hip_runtime.h>
#include <hip/hip_bf16.h>
#include <stdint.h>

// ---------------------------------------------------------------------------
// GNN pipeline for MI455X (gfx1250, wave32, WMMA).
// Big GEMM (400x160000 @ 160000x400) and MLP layers run on
// v_wmma_f32_16x16x32_bf16 with fp32 accumulation.
// ---------------------------------------------------------------------------

#define NNODES 400
#define KBIG   160000
#define NEG    0.01f

typedef __attribute__((ext_vector_type(16))) __bf16 v16bf;
typedef __attribute__((ext_vector_type(8)))  float  v8f;

struct __attribute__((aligned(16))) U128 { unsigned int x, y, z, w; };

__device__ __forceinline__ float lk(float x) { return x >= 0.f ? x : NEG * x; }

__device__ __forceinline__ unsigned short f2bf(float f) {
  unsigned u = __float_as_uint(f);
  u += 0x7FFFu + ((u >> 16) & 1u);          // round-to-nearest-even
  return (unsigned short)(u >> 16);
}

// ---------------------------- tiny graph kernels ---------------------------

__global__ void k_zero(float* p, int n) {
  int i = blockIdx.x * blockDim.x + threadIdx.x;
  if (i < n) p[i] = 0.f;
}

__global__ void k_degree(const int* __restrict__ src, const int* __restrict__ dst,
                         int E, float* degs, float* degd) {
  int e = blockIdx.x * blockDim.x + threadIdx.x;
  if (e < E) {
    atomicAdd(&degs[src[e]], 1.f);
    atomicAdd(&degd[dst[e]], 1.f);
  }
}

__global__ void k_norm(const float* degs, const float* degd, float* ns, float* nd, int n) {
  int i = blockIdx.x * blockDim.x + threadIdx.x;
  if (i < n) {
    ns[i] = 1.f / sqrtf(fmaxf(degs[i], 1.f));
    nd[i] = 1.f / sqrtf(fmaxf(degd[i], 1.f));
  }
}

// conv1 aggregate: agg1[dst] += features[src]*ns[src]   (400x3)
__global__ void k_scatter1(const float* __restrict__ feat, const int* __restrict__ src,
                           const int* __restrict__ dst, const float* __restrict__ ns,
                           float* agg1, int E) {
  int idx = blockIdx.x * blockDim.x + threadIdx.x;
  if (idx < E * 3) {
    int e = idx / 3, c = idx - e * 3;
    atomicAdd(&agg1[dst[e] * 3 + c], feat[src[e] * 3 + c] * ns[src[e]]);
  }
}

// x1 = leaky((agg1 @ W1) * nd + b1)   (400x5)
__global__ void k_conv1fin(const float* __restrict__ agg1, const float* __restrict__ W1,
                           const float* __restrict__ b1, const float* __restrict__ nd,
                           float* x1) {
  int idx = blockIdx.x * blockDim.x + threadIdx.x;
  if (idx < NNODES * 5) {
    int i = idx / 5, j = idx - i * 5;
    float s = 0.f;
    for (int c = 0; c < 3; ++c) s += agg1[i * 3 + c] * W1[c * 5 + j];
    x1[idx] = lk(s * nd[i] + b1[j]);
  }
}

// conv2 aggregate: A2[dst] += x1[src]*ns[src]   (400x5)
__global__ void k_scatter2(const float* __restrict__ x1, const int* __restrict__ src,
                           const int* __restrict__ dst, const float* __restrict__ ns,
                           float* A2, int E) {
  int idx = blockIdx.x * blockDim.x + threadIdx.x;
  if (idx < E * 5) {
    int e = idx / 5, j = idx - e * 5;
    atomicAdd(&A2[dst[e] * 5 + j], x1[src[e] * 5 + j] * ns[src[e]]);
  }
}

// B2 = A2 * nd[:,None]  (fold nd through positively-homogeneous leaky)
__global__ void k_B2(const float* A2, const float* nd, float* B2) {
  int idx = blockIdx.x * blockDim.x + threadIdx.x;
  if (idx < NNODES * 5) B2[idx] = A2[idx] * nd[idx / 5];
}

// X[i,k] = leaky(sum_p B2[i,p]*weight[p,k]) -> bf16, 8 elems/thread, coalesced
__global__ void k_buildX(const float* __restrict__ B2, const float* __restrict__ weight,
                         unsigned short* __restrict__ Xb) {
  int id = blockIdx.x * blockDim.x + threadIdx.x;      // 400 * 20000 threads
  if (id >= NNODES * (KBIG / 8)) return;
  int i = id / (KBIG / 8);
  int k = (id - i * (KBIG / 8)) * 8;
  float bv[5];
#pragma unroll
  for (int p = 0; p < 5; ++p) bv[p] = B2[i * 5 + p];
  float z[8];
#pragma unroll
  for (int u = 0; u < 8; ++u) z[u] = 0.f;
#pragma unroll
  for (int p = 0; p < 5; ++p) {
    const float4* wp = reinterpret_cast<const float4*>(weight + (size_t)p * KBIG + k);
    float4 w0 = wp[0], w1 = wp[1];
    float bp = bv[p];
    z[0] += bp * w0.x; z[1] += bp * w0.y; z[2] += bp * w0.z; z[3] += bp * w0.w;
    z[4] += bp * w1.x; z[5] += bp * w1.y; z[6] += bp * w1.z; z[7] += bp * w1.w;
  }
  union { unsigned short h[8]; U128 u; } o;
#pragma unroll
  for (int u = 0; u < 8; ++u) o.h[u] = f2bf(lk(z[u]));
  *reinterpret_cast<U128*>(Xb + (size_t)i * KBIG + k) = o.u;
}

// ------------------------------- WMMA GEMM ---------------------------------
// C(MxN) = A(bf16, MxKpad) @ B(fp32, KxN), optional bias+leaky or atomic-acc.
// Block = 128 threads = 4 waves; each wave owns one 16x16 C tile.
// CHUNK = 64: two v_wmma_f32_16x16x32_bf16 per chunk, two accumulators.
// Double-buffered LDS B slab -> ONE barrier per chunk; consume-before-fill so
// WMMAs never wait on the fill's loadcnt. Fast unguarded fill path when the
// chunk is fully in-bounds (always true for the big GEMM, K % 64 == 0).
// A read straight from global (per-lane k-sets are contiguous 8-elem runs);
// B staged coalesced into LDS, bf16-packed, [n][k] layout, 144B row stride
// (36-word stride -> 16 lanes hit 16 distinct banks, 16B aligned b128 reads).
#define GCHUNK 64
#define BSLAB  (16 * 72)                 // ushorts per buffer (2304 B)

__device__ __forceinline__ void gemm_fill(const float* __restrict__ B, int ldb,
                                          int jb, int K, int k0,
                                          unsigned short* ldsB, int t) {
  const int n  = t & 15;
  const int p0 = t >> 4;                                  // k-pair 0..7
  unsigned* lp = reinterpret_cast<unsigned*>(ldsB) + n * 36 + p0;
  if (k0 + GCHUNK <= K) {
    // fast path: no bounds checks, strength-reduced addressing
    const float* bp = B + (size_t)(k0 + 2 * p0) * ldb + jb + n;
    const size_t step = (size_t)16 * ldb;
#pragma unroll
    for (int j = 0; j < 4; ++j) {
      float v0 = bp[0];
      float v1 = bp[ldb];
      lp[j * 8] = (unsigned)f2bf(v0) | ((unsigned)f2bf(v1) << 16);
      bp += step;
    }
    // speculative prefetch of the chunk after this one
    int kn = k0 + GCHUNK;
    if (kn < K) {
      int r = t >> 1, c = (t & 1) * 8;
      if (kn + r < K) __builtin_prefetch(&B[(size_t)(kn + r) * ldb + jb + c], 0, 1);
    }
  } else {
    // tail path (only the last partial chunk of K=400 layers)
#pragma unroll
    for (int j = 0; j < 4; ++j) {
      int p = p0 + j * 8;
      int kg = k0 + 2 * p;
      float v0 = (kg     < K) ? B[(size_t)kg * ldb + jb + n] : 0.f;
      float v1 = (kg + 1 < K) ? B[(size_t)(kg + 1) * ldb + jb + n] : 0.f;
      lp[j * 8] = (unsigned)f2bf(v0) | ((unsigned)f2bf(v1) << 16);
    }
  }
}

__global__ void k_gemm(const unsigned short* __restrict__ A, int lda,
                       const float* __restrict__ B, int ldb,
                       float* __restrict__ C, int ldc,
                       int Mtiles, int Mrows, int K, int kspan,
                       const float* __restrict__ bias, int doLeaky, int doAtomic) {
  __shared__ __align__(16) unsigned short ldsB[2][BSLAB];  // double buffer

  const int t    = threadIdx.x;
  const int w    = t >> 5;
  const int lane = t & 31;
  const int m    = lane & 15;
  const int hf   = lane >> 4;
  const int rowTile = blockIdx.x * 4 + w;
  const bool active = rowTile < Mtiles;
  const int rt = active ? rowTile : 0;                     // clamp for addr math
  const int jb = blockIdx.y * 16;
  const int Kpad = (K + 63) & ~63;
  int ks = blockIdx.z * kspan;
  int ke = ks + kspan; if (ke > Kpad) ke = Kpad;
  const int nch = (ke > ks) ? (ke - ks + GCHUNK - 1) / GCHUNK : 0;

  const unsigned short* arow = A + (size_t)(rt * 16 + m) * (size_t)lda;

  v8f acc0 = {0.f, 0.f, 0.f, 0.f, 0.f, 0.f, 0.f, 0.f};
  v8f acc1 = {0.f, 0.f, 0.f, 0.f, 0.f, 0.f, 0.f, 0.f};

  if (nch) gemm_fill(B, ldb, jb, K, ks, ldsB[0], t);       // prologue fill

  if (active) {
    for (int c = 0; c < nch; ++c) {
      const int k0 = ks + c * GCHUNK;
      __syncthreads();                                     // publish buf[c&1]
      const char* brow = reinterpret_cast<const char*>(ldsB[c & 1]) + m * 144;
      // prefetch next chunk of A (two 64B lines across the two lane-halves)
      if (c + 1 < nch) __builtin_prefetch(arow + k0 + GCHUNK + hf * 32, 0, 1);
      // sub-chunk 0: k = k0 .. k0+31
      U128 a0 = *reinterpret_cast<const U128*>(arow + k0 + hf * 8);
      U128 a1 = *reinterpret_cast<const U128*>(arow + k0 + 16 + hf * 8);
      U128 b0 = *reinterpret_cast<const U128*>(brow + hf * 16);
      U128 b1 = *reinterpret_cast<const U128*>(brow + 32 + hf * 16);
      // sub-chunk 1: k = k0+32 .. k0+63
      U128 a2 = *reinterpret_cast<const U128*>(arow + k0 + 32 + hf * 8);
      U128 a3 = *reinterpret_cast<const U128*>(arow + k0 + 48 + hf * 8);
      U128 b2 = *reinterpret_cast<const U128*>(brow + 64 + hf * 16);
      U128 b3 = *reinterpret_cast<const U128*>(brow + 96 + hf * 16);
      union { U128 u[2]; v16bf v; } ua0, ub0, ua1, ub1;
      ua0.u[0] = a0; ua0.u[1] = a1;  ub0.u[0] = b0; ub0.u[1] = b1;
      ua1.u[0] = a2; ua1.u[1] = a3;  ub1.u[0] = b2; ub1.u[1] = b3;
      acc0 = __builtin_amdgcn_wmma_f32_16x16x32_bf16(
          false, ua0.v, false, ub0.v, (short)0, acc0, false, false);
      acc1 = __builtin_amdgcn_wmma_f32_16x16x32_bf16(
          false, ua1.v, false, ub1.v, (short)0, acc1, false, false);
      // fill the other buffer for the next chunk (overlaps WMMA execution)
      if (c + 1 < nch) gemm_fill(B, ldb, jb, K, k0 + GCHUNK, ldsB[(c + 1) & 1], t);
    }
  } else {
    // inactive waves: identical barrier schedule, still help fill the slab
    for (int c = 0; c < nch; ++c) {
      const int k0 = ks + c * GCHUNK;
      __syncthreads();
      if (c + 1 < nch) gemm_fill(B, ldb, jb, K, k0 + GCHUNK, ldsB[(c + 1) & 1], t);
    }
  }

  if (active) {
#pragma unroll
    for (int r = 0; r < 8; ++r) {
      int orow = rowTile * 16 + r + hf * 8;    // C/D layout: VGPR r -> M=r (+8 hi half)
      int ocol = jb + m;
      if (orow < Mrows) {
        float v = acc0[r] + acc1[r];
        if (bias)     v += bias[ocol];
        if (doLeaky)  v = lk(v);
        if (doAtomic) atomicAdd(&C[(size_t)orow * ldc + ocol], v);
        else          C[(size_t)orow * ldc + ocol] = v;
      }
    }
  }
}

// conv3 aggregate: y3agg[dst,:] += M[src,:]*ns[src]
__global__ void k_scatter3(const float* __restrict__ Mb, const int* __restrict__ src,
                           const int* __restrict__ dst, const float* __restrict__ ns,
                           float* y3agg, int E) {
  int idx = blockIdx.x * blockDim.x + threadIdx.x;
  if (idx < E * NNODES) {
    int e = idx / NNODES, j = idx - e * NNODES;
    atomicAdd(&y3agg[(size_t)dst[e] * NNODES + j],
              Mb[(size_t)src[e] * NNODES + j] * ns[src[e]]);
  }
}

__global__ void k_conv3fin(const float* __restrict__ y3agg, const float* __restrict__ nd,
                           const float* __restrict__ b3, float* y3) {
  int idx = blockIdx.x * blockDim.x + threadIdx.x;
  if (idx < NNODES * NNODES) {
    int i = idx / NNODES, j = idx - i * NNODES;
    y3[idx] = lk(y3agg[idx] * nd[i] + b3[j]);
  }
}

// fp32 -> bf16 with zero-padded K (Kpad multiple of 64)
__global__ void k_cvt(const float* __restrict__ src, unsigned short* __restrict__ dst,
                      int M, int K, int Kpad) {
  int idx = blockIdx.x * blockDim.x + threadIdx.x;
  if (idx < M * Kpad) {
    int r = idx / Kpad, k = idx - r * Kpad;
    float v = (k < K) ? src[(size_t)r * K + k] : 0.f;
    dst[idx] = f2bf(v);
  }
}

// final 64->1 layer, fp32
__global__ void k_final(const float* __restrict__ x7, const float* __restrict__ l5w,
                        const float* __restrict__ l5b, float* __restrict__ out) {
  int i = blockIdx.x * blockDim.x + threadIdx.x;
  if (i < NNODES) {
    float s = l5b[0];
    for (int j = 0; j < 64; ++j) s += x7[i * 64 + j] * l5w[j];
    out[i] = s;
  }
}

// ------------------------------- launcher ----------------------------------

extern "C" void kernel_launch(void* const* d_in, const int* in_sizes, int n_in,
                              void* d_out, int out_size, void* d_ws, size_t ws_size,
                              hipStream_t stream) {
  const float* features = (const float*)d_in[0];
  const int*   src      = (const int*)d_in[1];
  const int*   dst      = (const int*)d_in[2];
  const float* weight   = (const float*)d_in[3];
  const float* W1       = (const float*)d_in[4];
  const float* b1       = (const float*)d_in[5];
  const float* W3       = (const float*)d_in[6];
  const float* b3       = (const float*)d_in[7];
  const float* l1w = (const float*)d_in[8],  *l1b = (const float*)d_in[9];
  const float* l2w = (const float*)d_in[10], *l2b = (const float*)d_in[11];
  const float* l3w = (const float*)d_in[12], *l3b = (const float*)d_in[13];
  const float* l4w = (const float*)d_in[14], *l4b = (const float*)d_in[15];
  const float* l5w = (const float*)d_in[16], *l5b = (const float*)d_in[17];
  const int E = in_sizes[1];
  (void)n_in; (void)out_size; (void)ws_size;

  char* ws = (char*)d_ws;
  size_t off = 0;
  auto alloc = [&](size_t b) -> char* {
    char* p = ws + off;
    off = (off + b + 255) & ~(size_t)255;
    return p;
  };
  float* degs  = (float*)alloc(400 * 4);
  float* degd  = (float*)alloc(400 * 4);
  float* ns    = (float*)alloc(400 * 4);
  float* nd    = (float*)alloc(400 * 4);
  float* agg1  = (float*)alloc(1200 * 4);
  float* x1    = (float*)alloc(2000 * 4);
  float* A2    = (float*)alloc(2000 * 4);
  float* B2    = (float*)alloc(2000 * 4);
  unsigned short* Xb  = (unsigned short*)alloc((size_t)400 * KBIG * 2);   // 128 MB
  float* Mb    = (float*)alloc((size_t)160000 * 4);
  float* y3agg = (float*)alloc((size_t)160000 * 4);
  float* y3    = (float*)alloc((size_t)160000 * 4);
  unsigned short* y3b = (unsigned short*)alloc((size_t)400 * 448 * 2);
  float* x4    = (float*)alloc((size_t)400 * 512 * 4);
  unsigned short* x4b = (unsigned short*)alloc((size_t)400 * 512 * 2);
  float* x5    = (float*)alloc((size_t)400 * 1024 * 4);
  unsigned short* x5b = (unsigned short*)alloc((size_t)400 * 1024 * 2);
  float* x6    = (float*)alloc((size_t)400 * 512 * 4);
  unsigned short* x6b = (unsigned short*)alloc((size_t)400 * 512 * 2);
  float* x7    = (float*)alloc((size_t)400 * 64 * 4);

  auto g1 = [](int n) { return dim3((unsigned)((n + 255) / 256)); };

  // zero accumulators
  k_zero<<<g1(400),    256, 0, stream>>>(degs, 400);
  k_zero<<<g1(400),    256, 0, stream>>>(degd, 400);
  k_zero<<<g1(1200),   256, 0, stream>>>(agg1, 1200);
  k_zero<<<g1(2000),   256, 0, stream>>>(A2, 2000);
  k_zero<<<g1(160000), 256, 0, stream>>>(Mb, 160000);
  k_zero<<<g1(160000), 256, 0, stream>>>(y3agg, 160000);

  // degrees and norms
  k_degree<<<g1(E),   256, 0, stream>>>(src, dst, E, degs, degd);
  k_norm  <<<g1(400), 256, 0, stream>>>(degs, degd, ns, nd, 400);

  // conv1
  k_scatter1<<<g1(E * 3), 256, 0, stream>>>(features, src, dst, ns, agg1, E);
  k_conv1fin<<<g1(2000),  256, 0, stream>>>(agg1, W1, b1, nd, x1);

  // conv2 aggregation + fold nd
  k_scatter2<<<g1(E * 5), 256, 0, stream>>>(x1, src, dst, ns, A2, E);
  k_B2      <<<g1(2000),  256, 0, stream>>>(A2, nd, B2);

  // materialize X = leaky(B2 @ weight) in bf16  (400 x 160000)
  k_buildX<<<g1(400 * (KBIG / 8)), 256, 0, stream>>>(B2, weight, Xb);

  // big GEMM: Mb = X @ W3   (51.2 GFLOP, bf16 WMMA, K split 10x, atomic acc)
  k_gemm<<<dim3(7, 25, 10), 128, 0, stream>>>(Xb, KBIG, W3, 400, Mb, 400,
                                              25, 400, KBIG, 16000, nullptr, 0, 1);

  // conv3 aggregation + epilogue
  k_scatter3<<<g1(E * NNODES), 256, 0, stream>>>(Mb, src, dst, ns, y3agg, E);
  k_conv3fin<<<g1(160000),     256, 0, stream>>>(y3agg, nd, b3, y3);

  // MLP (bf16 WMMA, fused bias + leaky)
  k_cvt <<<g1(400 * 448),  256, 0, stream>>>(y3, y3b, 400, 400, 448);
  k_gemm<<<dim3(7, 32, 1), 128, 0, stream>>>(y3b, 448, l1w, 512, x4, 512,
                                             25, 400, 400, 448, l1b, 1, 0);
  k_cvt <<<g1(400 * 512),  256, 0, stream>>>(x4, x4b, 400, 512, 512);
  k_gemm<<<dim3(7, 64, 1), 128, 0, stream>>>(x4b, 512, l2w, 1024, x5, 1024,
                                             25, 400, 512, 512, l2b, 1, 0);
  k_cvt <<<g1(400 * 1024), 256, 0, stream>>>(x5, x5b, 400, 1024, 1024);
  k_gemm<<<dim3(7, 32, 1), 128, 0, stream>>>(x5b, 1024, l3w, 512, x6, 512,
                                             25, 400, 1024, 1024, l3b, 1, 0);
  k_cvt <<<g1(400 * 512),  256, 0, stream>>>(x6, x6b, 400, 512, 512);
  k_gemm<<<dim3(7, 4, 1),  128, 0, stream>>>(x6b, 512, l4w, 64, x7, 64,
                                             25, 400, 512, 512, l4b, 1, 0);

  k_final<<<g1(400), 256, 0, stream>>>(x7, l5w, l5b, (float*)d_out);
}
